// MoE_7791070675576
// MI455X (gfx1250) — matrix operand order repaired and checked by reading, not compiled, not run on hardware
//
#include <hip/hip_runtime.h>
#include <stdint.h>

// ---------------------------------------------------------------- types
typedef __attribute__((ext_vector_type(16))) __bf16 v16bf;
typedef __attribute__((ext_vector_type(8)))  float  v8f;

#define NTOK 8192
#define DIM  1024
#define HID  4096   // 4*DIM
#define GHID 2048   // 2*DIM
#define NEXP 8

// block tile: BM=128 rows x BN=256 cols, BK=32; 8 wave32s, each 64x64
#define AP_STRIDE 20    // uints per A-LDS row: 16 k-pairs + 4 pad
#define BP_STRIDE 260   // uints per B-LDS row: 256 cols + 4 pad
#define AP_SZ (128 * AP_STRIDE)
#define BP_SZ (16 * BP_STRIDE)

// ---------------------------------------------------------------- helpers
__device__ __forceinline__ uint32_t pack_bf16(float a, float b) {
  union { __bf16 h[2]; uint32_t u; } r;
  r.h[0] = (__bf16)a;            // native v_cvt bf16 (RNE)
  r.h[1] = (__bf16)b;
  return r.u;
}
__device__ __forceinline__ float bf16_bits_to_f32(unsigned short b) {
  return __uint_as_float(((uint32_t)b) << 16);
}

// A fragment (16x32 bf16, MxK). Lane L: m = mbase + (L&15); VGPR i<4 holds
// K = 2i,2i+1 (+8 for upper lane half); VGPR 4..7 holds K = 16+2(i-4)(,+1)(+8).
__device__ __forceinline__ v16bf frag_a(const uint32_t* APb, int mbase, int lane) {
  const uint32_t* row = APb + (mbase + (lane & 15)) * AP_STRIDE;
  const int pb = (lane >> 4) * 4;
  union { uint32_t u[8]; v16bf v; } f;
#pragma unroll
  for (int i = 0; i < 4; ++i) {
    f.u[i]     = row[pb + i];
    f.u[4 + i] = row[8 + pb + i];
  }
  return f.v;
}

// B fragment (32x16 bf16, KxN). Lane L: n = nbase + (L&15); VGPR j<4 holds
// K = 2j,2j+1 (+8 for upper lane half); VGPR 4..7 holds K = 16+2(j-4)(,+1)(+8).
__device__ __forceinline__ v16bf frag_b(const uint32_t* BPb, int nbase, int lane) {
  const int n  = nbase + (lane & 15);
  const int pb = (lane >> 4) * 4;
  union { uint32_t u[8]; v16bf v; } f;
#pragma unroll
  for (int j = 0; j < 8; ++j) f.u[j] = BPb[(pb + (j < 4 ? j : j + 4)) * BP_STRIDE + n];
  return f.v;
}

// stage one 128x32 f32 A tile -> pair-packed bf16 LDS
__device__ __forceinline__ void stage_A_f32(const float* __restrict__ A,
                                            uint32_t* APb, int rowBlk, int K,
                                            int k0, int arow, int aseg) {
  const float4* ga = (const float4*)(A + (size_t)(rowBlk + arow) * K + k0 + aseg);
  float4 a0 = ga[0], a1 = ga[1], a2 = ga[2], a3 = ga[3];
  uint32_t* ap = APb + arow * AP_STRIDE + (aseg >> 1);
  ap[0] = pack_bf16(a0.x, a0.y); ap[1] = pack_bf16(a0.z, a0.w);
  ap[2] = pack_bf16(a1.x, a1.y); ap[3] = pack_bf16(a1.z, a1.w);
  ap[4] = pack_bf16(a2.x, a2.y); ap[5] = pack_bf16(a2.z, a2.w);
  ap[6] = pack_bf16(a3.x, a3.y); ap[7] = pack_bf16(a3.z, a3.w);
}

// stage one 32x256 f32 B tile -> k-pair-packed bf16 LDS (lo=even k, hi=odd k)
__device__ __forceinline__ void stage_B_f32(const float* __restrict__ gb0, int ldb,
                                            uint32_t* BPb, int bp, int bseg) {
  const float* gb1 = gb0 + ldb;
  uint32_t* bq = BPb + bp * BP_STRIDE + bseg;
#pragma unroll
  for (int q = 0; q < 4; ++q) {
    float4 r0 = ((const float4*)gb0)[q];
    float4 r1 = ((const float4*)gb1)[q];
    bq[q * 4 + 0] = pack_bf16(r0.x, r1.x);
    bq[q * 4 + 1] = pack_bf16(r0.y, r1.y);
    bq[q * 4 + 2] = pack_bf16(r0.z, r1.z);
    bq[q * 4 + 3] = pack_bf16(r0.w, r1.w);
  }
}

// 64x64 wave tile: 16 WMMAs against one staged LDS buffer
__device__ __forceinline__ void compute_tile(const uint32_t* APb, const uint32_t* BPb,
                                             int waveM, int waveN, int lane,
                                             v8f (&acc)[4][4]) {
  v16bf bfr[4];
#pragma unroll
  for (int ni = 0; ni < 4; ++ni)
    bfr[ni] = frag_b(BPb, waveN * 64 + ni * 16, lane);
#pragma unroll
  for (int mi = 0; mi < 4; ++mi) {
    v16bf af = frag_a(APb, waveM * 64 + mi * 16, lane);
#pragma unroll
    for (int ni = 0; ni < 4; ++ni)
      acc[mi][ni] = __builtin_amdgcn_wmma_f32_16x16x32_bf16(
          false, af, false, bfr[ni], (short)0, acc[mi][ni], false, false);
  }
}

// ---------------------------------------------------------------- zero init
__global__ void moe_zero_kernel(float* __restrict__ out, int* __restrict__ cnt, int total) {
  int i = blockIdx.x * 256 + threadIdx.x;
  if (i < total) out[i] = 0.0f;
  if (blockIdx.x == 0 && threadIdx.x < NEXP) cnt[threadIdx.x] = 0;
}

// ---------------------------------------------------------------- GEMM + SiLU
// C(bf16)[M,N] = silu(A(f32)[M,K] @ B(f32)[K,N] + bias[N]); double-buffered LDS
__global__ __launch_bounds__(256) void gemm_silu_kernel(
    const float* __restrict__ A, const float* __restrict__ B,
    const float* __restrict__ bias, __bf16* __restrict__ C,
    int M, int N, int K)
{
  __shared__ uint32_t AP[2][AP_SZ];
  __shared__ uint32_t BP[2][BP_SZ];

  const int t = threadIdx.x, lane = t & 31, wave = t >> 5;
  const int waveM = wave & 1, waveN = wave >> 1;       // 2 x 4 waves, 64x64 each
  const int rowBlk = blockIdx.y * 128, colBlk = blockIdx.x * 256;

  v8f acc[4][4];
#pragma unroll
  for (int mi = 0; mi < 4; ++mi)
#pragma unroll
    for (int ni = 0; ni < 4; ++ni)
#pragma unroll
      for (int v = 0; v < 8; ++v) acc[mi][ni][v] = 0.0f;

  const int arow = t >> 1, aseg = (t & 1) * 16;  // A: 16 f32 along K
  const int bp   = t >> 4, bseg = (t & 15) * 16; // B: 16 cols x 2 k-rows
  const float* gbBase = B + (size_t)(2 * bp) * N + colBlk + bseg;

  // prologue: stage tile 0
  stage_A_f32(A, AP[0], rowBlk, K, 0, arow, aseg);
  stage_B_f32(gbBase, N, BP[0], bp, bseg);

  const int nk = K / 32;
  for (int ik = 0; ik < nk; ++ik) {
    const int buf = ik & 1;
    __syncthreads();                       // staged tile[buf] visible; buf^1 free
    if (ik + 1 < nk) {                     // stage next tile into other buffer
      const int k1 = (ik + 1) * 32;
      stage_A_f32(A, AP[buf ^ 1], rowBlk, K, k1, arow, aseg);
      stage_B_f32(gbBase + (size_t)k1 * N, N, BP[buf ^ 1], bp, bseg);
      if (ik + 2 < nk)
        __builtin_prefetch(gbBase + (size_t)(k1 + 32) * N, 0, 0);
    }
    compute_tile(AP[buf], BP[buf], waveM, waveN, lane, acc);
  }

  // epilogue: bias + SiLU -> bf16
  const int halfAdd = (lane >> 4) * 8;
  const int nlane   = lane & 15;
#pragma unroll
  for (int mi = 0; mi < 4; ++mi)
#pragma unroll
    for (int ni = 0; ni < 4; ++ni) {
      int col = colBlk + waveN * 64 + ni * 16 + nlane;
      float bb = bias[col];
#pragma unroll
      for (int v = 0; v < 8; ++v) {
        int row = rowBlk + waveM * 64 + mi * 16 + halfAdd + v;
        float x = acc[mi][ni][v] + bb;
        x = x / (1.0f + __expf(-x));      // SiLU
        C[(size_t)row * N + col] = (__bf16)x;
      }
    }
}

// ---------------------------------------------------------------- gate layer 2 + top-2
__global__ __launch_bounds__(256) void gate_topk_kernel(
    const unsigned short* __restrict__ G1,  // [NTOK, GHID] bf16
    const float* __restrict__ Wg2,          // [GHID, NEXP]
    const float* __restrict__ bg2,          // [NEXP]
    int* __restrict__ idx, float* __restrict__ wts)
{
  const int lane = threadIdx.x & 31;
  const int n = blockIdx.x * 8 + (threadIdx.x >> 5);

  float acc[NEXP];
#pragma unroll
  for (int e = 0; e < NEXP; ++e) acc[e] = 0.0f;

  const unsigned short* g = G1 + (size_t)n * GHID + lane * 64;
  const float* wg = Wg2 + (size_t)lane * 64 * NEXP;
  for (int j = 0; j < 64; ++j) {
    float a = bf16_bits_to_f32(g[j]);
    const float4* wr = (const float4*)(wg + j * NEXP);
    float4 w0 = wr[0], w1 = wr[1];
    acc[0] += a * w0.x; acc[1] += a * w0.y; acc[2] += a * w0.z; acc[3] += a * w0.w;
    acc[4] += a * w1.x; acc[5] += a * w1.y; acc[6] += a * w1.z; acc[7] += a * w1.w;
  }
#pragma unroll
  for (int off = 16; off > 0; off >>= 1)
#pragma unroll
    for (int e = 0; e < NEXP; ++e) acc[e] += __shfl_xor(acc[e], off, 32);

  if (lane == 0) {
#pragma unroll
    for (int e = 0; e < NEXP; ++e) acc[e] += bg2[e];
    int i0 = 0;
#pragma unroll
    for (int e = 1; e < NEXP; ++e) if (acc[e] > acc[i0]) i0 = e;
    int i1 = (i0 == 0) ? 1 : 0;
#pragma unroll
    for (int e = 0; e < NEXP; ++e)
      if (e != i0 && acc[e] > acc[i1]) i1 = e;
    float ex = __expf(acc[i1] - acc[i0]);   // g1 <= g0
    float inv = 1.0f / (1.0f + ex);
    idx[n * 2 + 0] = i0; idx[n * 2 + 1] = i1;
    wts[n * 2 + 0] = inv; wts[n * 2 + 1] = ex * inv;
  }
}

// ---------------------------------------------------------------- expert compaction
__global__ void scatter_kernel(const int* __restrict__ idx,
                               int* __restrict__ cnt, int* __restrict__ perm) {
  int n = blockIdx.x * 256 + threadIdx.x;
#pragma unroll
  for (int s = 0; s < 2; ++s) {
    int e = idx[n * 2 + s];
    int pos = atomicAdd(&cnt[e], 1);
    perm[e * NTOK + pos] = (n << 1) | s;
  }
}

// ---------------------------------------------------------------- gathered expert ff2
// out[token] += w * (h[token] @ W2[e] + b2[e]) for tokens routed to expert e.
// Gathered A rows (bf16) streamed with CDNA5 async LDS loads, double-buffered
// so the async DMA overlaps the WMMA pipe; wait deferred to next iteration.
__global__ __launch_bounds__(256) void expert_ff2_kernel(
    const unsigned short* __restrict__ H,   // [NTOK, HID] bf16
    const float* __restrict__ W2,           // [NEXP, HID, DIM]
    const float* __restrict__ b2,           // [NEXP, DIM]
    const int*   __restrict__ perm,         // [NEXP, NTOK]
    const int*   __restrict__ cnt,          // [NEXP]
    const float* __restrict__ wts,          // [NTOK, 2]
    float* __restrict__ out)                // [NTOK, DIM]
{
  const int e = blockIdx.z;
  const int rows = cnt[e];
  const int rowBlk = blockIdx.y * 128;
  if (rowBlk >= rows) return;
  const int colBlk = blockIdx.x * 256;

  __shared__ uint32_t AP[2][AP_SZ];
  __shared__ uint32_t BP[2][BP_SZ];
  __shared__ int codeS[128];

  const int t = threadIdx.x, lane = t & 31, wave = t >> 5;
  const int waveM = wave & 1, waveN = wave >> 1;

  if (t < 128) {
    int r = rowBlk + t;
    codeS[t] = (r < rows) ? perm[e * NTOK + r] : -1;
  }
  __syncthreads();

  v8f acc[4][4];
#pragma unroll
  for (int mi = 0; mi < 4; ++mi)
#pragma unroll
    for (int ni = 0; ni < 4; ++ni)
#pragma unroll
      for (int v = 0; v < 8; ++v) acc[mi][ni][v] = 0.0f;

  const int arow = t >> 1, aseg = (t & 1);       // each thread: 32B of one row
  const int bp   = t >> 4, bseg = (t & 15) * 16;
  int codeMine = codeS[arow];
  const char* hrow = (const char*)H +
      (size_t)(codeMine >= 0 ? (codeMine >> 1) : 0) * (HID * 2);
  const uint32_t ldsA[2] = {
      (uint32_t)(uintptr_t)(&AP[0][0] + arow * AP_STRIDE + aseg * 8),
      (uint32_t)(uintptr_t)(&AP[1][0] + arow * AP_STRIDE + aseg * 8)};
  const float* W2e = W2 + (size_t)e * HID * DIM;
  const float* gbBase = W2e + (size_t)(2 * bp) * DIM + colBlk + bseg;

  // issue async A-stage for k-tile into LDS buffer `b`
  auto issue_async_A = [&](int b, int k0) {
    uint64_t ga = (uint64_t)(uintptr_t)(hrow + k0 * 2 + aseg * 32);
    uint32_t la = ldsA[b];
    asm volatile("global_load_async_to_lds_b128 %0, %1, off"
                 :: "v"(la), "v"(ga) : "memory");
    uint64_t ga2 = ga + 16;
    uint32_t la2 = la + 16;
    asm volatile("global_load_async_to_lds_b128 %0, %1, off"
                 :: "v"(la2), "v"(ga2) : "memory");
  };

  // prologue: stage tile 0
  issue_async_A(0, 0);
  stage_B_f32(gbBase, DIM, BP[0], bp, bseg);

  const int nk = HID / 32;
  for (int ik = 0; ik < nk; ++ik) {
    const int buf = ik & 1;
    asm volatile("s_wait_asynccnt 0" ::: "memory");  // A[buf] DMA complete
    __syncthreads();                                 // tile[buf] visible to all
    if (ik + 1 < nk) {                               // overlap next-tile staging
      const int k1 = (ik + 1) * 32;
      issue_async_A(buf ^ 1, k1);
      stage_B_f32(gbBase + (size_t)k1 * DIM, DIM, BP[buf ^ 1], bp, bseg);
      if (ik + 2 < nk)
        __builtin_prefetch(gbBase + (size_t)(k1 + 32) * DIM, 0, 0);
    }
    compute_tile(AP[buf], BP[buf], waveM, waveN, lane, acc);
  }

  // epilogue: (acc + b2) * gate_weight, atomic combine (2 commuting adds/elt)
  const int halfAdd = (lane >> 4) * 8;
  const int nlane   = lane & 15;
#pragma unroll
  for (int mi = 0; mi < 4; ++mi)
#pragma unroll
    for (int ni = 0; ni < 4; ++ni) {
      int col = colBlk + waveN * 64 + ni * 16 + nlane;
      float bb = b2[e * DIM + col];
#pragma unroll
      for (int v = 0; v < 8; ++v) {
        int lr = waveM * 64 + mi * 16 + halfAdd + v;
        int code = codeS[lr];
        if (code >= 0) {
          int token = code >> 1;
          float wv = wts[token * 2 + (code & 1)];
          float val = (acc[mi][ni][v] + bb) * wv;
          __hip_atomic_fetch_add(&out[(size_t)token * DIM + col], val,
                                 __ATOMIC_RELAXED, __HIP_MEMORY_SCOPE_AGENT);
        }
      }
    }
}

// ---------------------------------------------------------------- launcher
extern "C" void kernel_launch(void* const* d_in, const int* in_sizes, int n_in,
                              void* d_out, int out_size, void* d_ws, size_t ws_size,
                              hipStream_t stream) {
  const float* x   = (const float*)d_in[0];
  const float* Wg1 = (const float*)d_in[1];
  const float* bg1 = (const float*)d_in[2];
  const float* Wg2 = (const float*)d_in[3];
  const float* bg2 = (const float*)d_in[4];
  const float* W1  = (const float*)d_in[5];
  const float* b1  = (const float*)d_in[6];
  const float* W2  = (const float*)d_in[7];
  const float* b2  = (const float*)d_in[8];
  float* out = (float*)d_out;

  // workspace layout (bytes)
  char* ws = (char*)d_ws;
  __bf16* g1 = (__bf16*)(ws);                                    // 32 MB
  __bf16* h  = (__bf16*)(ws + (size_t)NTOK * GHID * 2);          // 64 MB
  char* p = ws + (size_t)NTOK * GHID * 2 + (size_t)NTOK * HID * 2;
  int*   idx  = (int*)p;            p += (size_t)NTOK * 2 * 4;
  float* wts  = (float*)p;          p += (size_t)NTOK * 2 * 4;
  int*   cnt  = (int*)p;            p += 256;
  int*   perm = (int*)p;

  moe_zero_kernel<<<(NTOK * DIM + 255) / 256, 256, 0, stream>>>(out, cnt, NTOK * DIM);

  // g1 = silu(x @ Wg1 + bg1)   [8192, 2048]
  gemm_silu_kernel<<<dim3(GHID / 256, NTOK / 128), 256, 0, stream>>>(
      x, Wg1, bg1, g1, NTOK, GHID, DIM);
  // h = silu(x @ W1 + b1)      [8192, 4096]
  gemm_silu_kernel<<<dim3(HID / 256, NTOK / 128), 256, 0, stream>>>(
      x, W1, b1, h, NTOK, HID, DIM);
  // gates -> top2 softmax
  gate_topk_kernel<<<NTOK / 8, 256, 0, stream>>>((const unsigned short*)g1,
                                                 Wg2, bg2, idx, wts);
  // compact token lists per expert
  scatter_kernel<<<NTOK / 256, 256, 0, stream>>>(idx, cnt, perm);
  // gathered expert GEMM + weighted combine
  expert_ff2_kernel<<<dim3(DIM / 256, NTOK / 128, NEXP), 256, 0, stream>>>(
      (const unsigned short*)h, W2, b2, perm, cnt, wts, out);
}
